// Attention_55757265436862
// MI455X (gfx1250) — matrix operand order repaired and checked by reading, not compiled, not run on hardware
//
#include <hip/hip_runtime.h>

typedef __attribute__((ext_vector_type(16))) _Float16 v16h;
typedef __attribute__((ext_vector_type(8)))  _Float16 v8h;
typedef __attribute__((ext_vector_type(8)))  float    v8f;

#define HEADS 4
#define DIM   256
#define HC    64
#define NPIX  2304   // 48*48
#define BATCH 4

// ---- WMMA helpers -----------------------------------------------------------

static __device__ __forceinline__ v8f wmma_f16(v16h a, v16h b, v8f c) {
  // D = A(16x32 f16) * B(32x16 f16) + C(16x16 f32)
  return __builtin_amdgcn_wmma_f32_16x16x32_f16(
      /*neg_a=*/false, a, /*neg_b=*/false, b,
      /*c_mod=*/(short)0, c, /*reuse_a=*/false, /*reuse_b=*/false);
}

// A-operand per-lane gather (16-bit A 16x32):
//   lanes 0-15  hold K = {0..7, 16..23}
//   lanes 16-31 hold K = {8..15, 24..31}
// base points at the M-row start of the 32-K chunk (contiguous K in memory).
static __device__ __forceinline__ v16h load_a16(const _Float16* base, bool hi) {
  v8h lo  = *(const v8h*)(base + (hi ? 8 : 0));
  v8h hi8 = *(const v8h*)(base + 16 + (hi ? 8 : 0));
  return __builtin_shufflevector(lo, hi8, 0,1,2,3,4,5,6,7,8,9,10,11,12,13,14,15);
}

// B-operand per-lane gather (16-bit B 32x16), same K order as A:
// lanes 0-15 -> K {0..7, 16..23}, lanes 16-31 -> K {8..15, 24..31}.
static __device__ __forceinline__ v16h load_b16(const _Float16* base, bool hi) {
  v8h lo  = *(const v8h*)(base + (hi ? 8 : 0));
  v8h hi8 = *(const v8h*)(base + 16 + (hi ? 8 : 0));
  return __builtin_shufflevector(lo, hi8, 0,1,2,3,4,5,6,7,8,9,10,11,12,13,14,15);
}

// Async global->LDS copy of 16 bytes (gfx1250 GLOBAL_LOAD_ASYNC_TO_LDS_B128).
// ldsoff = wave-relative LDS byte offset (low 32 bits of the flat LDS address).
static __device__ __forceinline__ void async_copy_b128(unsigned ldsoff,
                                                       unsigned long long gaddr) {
  asm volatile("global_load_async_to_lds_b128 %0, %1, off"
               :: "v"(ldsoff), "v"(gaddr) : "memory");
}
static __device__ __forceinline__ void wait_asynccnt0() {
  asm volatile("s_wait_asynccnt 0" ::: "memory");
}
static __device__ __forceinline__ unsigned lds_off(const void* p) {
  return (unsigned)(unsigned long long)(uintptr_t)p;   // LDS aperture: addr[31:0]
}

// ---- Stage 0: f32 -> f16 conversions ---------------------------------------

// x [b,C,N] f32  ->  xT [b,N,C] f16  (B-operand friendly)
__global__ void cvt_x_kernel(const float* __restrict__ x, _Float16* __restrict__ xT) {
  int idx = blockIdx.x * blockDim.x + threadIdx.x;        // c fastest
  int c = idx & (DIM - 1);
  int rem = idx / DIM;
  int n = rem % NPIX;
  int b = rem / NPIX;
  xT[idx] = (_Float16)x[((size_t)b * DIM + c) * NPIX + n];
}

// wk,wq,wv,wo [256,256] f32 -> wh (4 contiguous f16 matrices, row-major [o][c])
__global__ void cvt_w_kernel(const float* __restrict__ wk, const float* __restrict__ wq,
                             const float* __restrict__ wv, const float* __restrict__ wo,
                             _Float16* __restrict__ wh) {
  int idx = blockIdx.x * blockDim.x + threadIdx.x;        // 0 .. 65535
  wh[idx]             = (_Float16)wk[idx];
  wh[idx + 65536]     = (_Float16)wq[idx];
  wh[idx + 2*65536]   = (_Float16)wv[idx];
  wh[idx + 3*65536]   = (_Float16)wo[idx];
}

// ---- Stage 1: Q/K/V projections (WMMA GEMM 256x256x9216 each) ---------------
// y[b,o,n] = sum_c W[o,c] * x[b,c,n] + bias[o]
// p=0 -> kT[b,h,n,hc], p=1 -> qT[b,h,n,hc], p=2 -> v[b,h,hc,n]
__global__ void proj_kernel(const _Float16* __restrict__ xT,
                            const _Float16* __restrict__ wh,
                            const float* __restrict__ bk,
                            const float* __restrict__ bq,
                            const float* __restrict__ bv,
                            _Float16* __restrict__ kT,
                            _Float16* __restrict__ qT,
                            _Float16* __restrict__ vN) {
  const int wave = threadIdx.x >> 5;
  const int lane = threadIdx.x & 31;
  const int l    = lane & 15;
  const bool hi  = lane >= 16;

  int t = blockIdx.x * 8 + wave;          // 27648 tiles total
  const int nt = t % 144; t /= 144;       // n tile (16 cols)
  const int ot = t % 16;  t /= 16;        // o tile (16 rows)
  const int b  = t % 4;
  const int p  = t / 4;                   // 0=k 1=q 2=v

  const _Float16* W    = wh + (size_t)p * DIM * DIM;
  const float*    bias = (p == 0) ? bk : (p == 1) ? bq : bv;
  const int o0 = ot * 16, n0 = nt * 16;

  const _Float16* arow = W  + (size_t)(o0 + l) * DIM;
  const _Float16* brow = xT + ((size_t)b * NPIX + (n0 + l)) * DIM;

  v8f acc = {};
  #pragma unroll
  for (int kk = 0; kk < 8; ++kk)          // K = 256 = 8 * 32
    acc = wmma_f16(load_a16(arow + kk * 32, hi), load_b16(brow + kk * 32, hi), acc);

  const int orow0 = o0 + (hi ? 8 : 0);    // first of this lane's 8 output rows
  #pragma unroll
  for (int r = 0; r < 8; ++r) acc[r] += bias[orow0 + r];

  const int h  = o0 / HC;                 // 16-row tile never crosses a head
  const int c0 = orow0 % HC;
  const int n  = n0 + l;                  // lane's output column

  if (p == 2) {
    // v natural [b,h,hc,N] : 8 strided f16 stores
    _Float16* dst = vN + (size_t)(b * HEADS + h) * HC * NPIX;
    #pragma unroll
    for (int r = 0; r < 8; ++r)
      dst[(size_t)(c0 + r) * NPIX + n] = (_Float16)acc[r];
  } else {
    // transposed [b,h,N,hc] : one 16B store (8 contiguous channels)
    _Float16* dst = (p == 0 ? kT : qT) +
                    ((size_t)(b * HEADS + h) * NPIX + n) * HC + c0;
    v8h hv;
    #pragma unroll
    for (int r = 0; r < 8; ++r) hv[r] = (_Float16)acc[r];
    *(v8h*)dst = hv;
  }
}

// ---- Stage 2: flash attention per (b,h) -------------------------------------
// S[n,m] = sum_c K[c,n] Q[c,m]; softmax over n; O[c,m] = sum_n V[c,n] P[n,m]
// One wave owns 32 queries (2 x 16-col tiles); 4 waves per block (128 queries).
// K/V tiles for each 32-key chunk are staged once per block into LDS with
// async global->LDS copies (ASYNCcnt), shared by all 4 waves.
__global__ void attn_kernel(const _Float16* __restrict__ qT,
                            const _Float16* __restrict__ kT,
                            const _Float16* __restrict__ vN,
                            _Float16* __restrict__ outT) {
  __shared__ _Float16 kbuf[32 * HC];        // K rows n0..n0+31, 64 halfs each (4 KB)
  __shared__ _Float16 vbuf[HC * 32];        // V rows c 0..63, 32 halfs each  (4 KB)
  __shared__ _Float16 pbuf[4][2][16 * 32];  // per-wave, per-m-subtile P      (8 KB)

  const int tid  = threadIdx.x;
  const int wave = tid >> 5;
  const int lane = tid & 31;
  const int l    = lane & 15;
  const bool hi  = lane >= 16;

  const int bh   = blockIdx.x / 18;       // b*HEADS + h
  const int mblk = blockIdx.x % 18;
  const int b    = bh / HEADS;
  const int h    = bh % HEADS;
  const int m0   = mblk * 128 + wave * 32;

  const _Float16* qTh = qT + (size_t)bh * NPIX * HC;
  const _Float16* kTh = kT + (size_t)bh * NPIX * HC;
  const _Float16* vh  = vN + (size_t)bh * HC * NPIX;

  // Q B-operands: invariant over the key loop (2 m-subtiles x 2 c-chunks)
  v16h bq[2][2];
  #pragma unroll
  for (int ms = 0; ms < 2; ++ms) {
    const int m = m0 + ms * 16 + l;
    bq[ms][0] = load_b16(qTh + (size_t)m * HC, hi);
    bq[ms][1] = load_b16(qTh + (size_t)m * HC + 32, hi);
  }

  // cooperative async-staging addresses for this thread
  const int krow = tid >> 2, kq = tid & 3;        // 32 rows x 128B, 32B per lane
  const int vrow = tid >> 1, vq = tid & 1;        // 64 rows x  64B, 32B per lane
  const unsigned kl0 = lds_off(&kbuf[krow * HC + kq * 16]);
  const unsigned vl0 = lds_off(&vbuf[vrow * 32 + vq * 16]);

  float m_run[2] = {-1e30f, -1e30f};
  float l_run[2] = {0.0f, 0.0f};
  v8f oa[2][4] = {};

  for (int n0 = 0; n0 < NPIX; n0 += 32) {
    __syncthreads();                      // staging buffers free to overwrite

    // ---- async global -> LDS staging of the shared K and V tiles
    unsigned long long kg =
        (unsigned long long)(uintptr_t)(kTh + (size_t)(n0 + krow) * HC + kq * 16);
    unsigned long long vg =
        (unsigned long long)(uintptr_t)(vh + (size_t)vrow * NPIX + n0 + vq * 16);
    async_copy_b128(kl0,      kg);
    async_copy_b128(kl0 + 16, kg + 16);
    async_copy_b128(vl0,      vg);
    async_copy_b128(vl0 + 16, vg + 16);
    wait_asynccnt0();
    __syncthreads();                      // staged data visible to all waves

    // ---- V A-operands from LDS (shared across both m-subtiles)
    v16h av[4];
    #pragma unroll
    for (int a = 0; a < 4; ++a)
      av[a] = load_a16(&vbuf[(a * 16 + l) * 32], hi);

    // ---- K A-operands from LDS
    const v16h ak00 = load_a16(&kbuf[(l)      * HC], hi);
    const v16h ak01 = load_a16(&kbuf[(l)      * HC + 32], hi);
    const v16h ak10 = load_a16(&kbuf[(16 + l) * HC], hi);
    const v16h ak11 = load_a16(&kbuf[(16 + l) * HC + 32], hi);

    #pragma unroll
    for (int ms = 0; ms < 2; ++ms) {
      // ---- S = K^T Q, two 16-row n-subtiles, K-dim = 64 channels
      v8f s0 = {}, s1 = {};
      s0 = wmma_f16(ak00, bq[ms][0], s0);
      s0 = wmma_f16(ak01, bq[ms][1], s0);
      s1 = wmma_f16(ak10, bq[ms][0], s1);
      s1 = wmma_f16(ak11, bq[ms][1], s1);

      // ---- online softmax over this 32-key chunk (lanes l / l+16 = one col)
      float cmax = -1e30f;
      #pragma unroll
      for (int r = 0; r < 8; ++r) { cmax = fmaxf(cmax, s0[r]); cmax = fmaxf(cmax, s1[r]); }
      cmax = fmaxf(cmax, __shfl_xor(cmax, 16, 32));
      const float m_new = fmaxf(m_run[ms], cmax);
      const float alpha = __expf(m_run[ms] - m_new);

      float psum = 0.0f;
      v8h p0h, p1h;
      #pragma unroll
      for (int r = 0; r < 8; ++r) {
        const float p0 = __expf(s0[r] - m_new);
        const float p1 = __expf(s1[r] - m_new);
        psum += p0 + p1;
        p0h[r] = (_Float16)p0;
        p1h[r] = (_Float16)p1;
      }
      psum += __shfl_xor(psum, 16, 32);
      l_run[ms] = l_run[ms] * alpha + psum;
      m_run[ms] = m_new;

      // ---- D-layout -> B-layout reshape of P through per-wave LDS
      _Float16* lds = pbuf[wave][ms];
      *(v8h*)(lds + l * 32 + (hi ? 8 : 0))      = p0h;   // n 0..15
      *(v8h*)(lds + l * 32 + 16 + (hi ? 8 : 0)) = p1h;   // n 16..31
      const v16h bp = load_b16(lds + l * 32, hi);        // same-wave DS: in order

      // ---- O = O*alpha + V * P   (hc = 64 = 4 x 16 M-rows, K = 32 keys)
      #pragma unroll
      for (int a = 0; a < 4; ++a) {
        #pragma unroll
        for (int r = 0; r < 8; ++r) oa[ms][a][r] *= alpha;
        oa[ms][a] = wmma_f16(av[a], bp, oa[ms][a]);
      }
    }
  }

  // ---- normalize and store outT[b, m, h*64 + c] (f16, B-operand friendly)
  const int cbase = hi ? 8 : 0;
  #pragma unroll
  for (int ms = 0; ms < 2; ++ms) {
    const float inv = 1.0f / l_run[ms];
    const int m = m0 + ms * 16 + l;
    _Float16* dst = outT + ((size_t)b * NPIX + m) * DIM + h * HC;
    #pragma unroll
    for (int a = 0; a < 4; ++a) {
      v8h hv;
      #pragma unroll
      for (int r = 0; r < 8; ++r) hv[r] = (_Float16)(oa[ms][a][r] * inv);
      *(v8h*)(dst + a * 16 + cbase) = hv;
    }
  }
}

// ---- Stage 3: output projection (WMMA GEMM) + bias, fp32 result -------------
__global__ void outproj_kernel(const _Float16* __restrict__ wh,   // wo at slot 3
                               const _Float16* __restrict__ outT,
                               const float* __restrict__ bo,
                               float* __restrict__ y) {
  const int wave = threadIdx.x >> 5;
  const int lane = threadIdx.x & 31;
  const int l    = lane & 15;
  const bool hi  = lane >= 16;

  int t = blockIdx.x * 8 + wave;          // 9216 tiles
  const int nt = t % 144; t /= 144;
  const int ot = t % 16;
  const int b  = t / 16;
  const int o0 = ot * 16, n0 = nt * 16;

  const _Float16* woh  = wh + (size_t)3 * DIM * DIM;
  const _Float16* arow = woh  + (size_t)(o0 + l) * DIM;
  const _Float16* brow = outT + ((size_t)b * NPIX + (n0 + l)) * DIM;

  v8f acc = {};
  #pragma unroll
  for (int kk = 0; kk < 8; ++kk)
    acc = wmma_f16(load_a16(arow + kk * 32, hi), load_b16(brow + kk * 32, hi), acc);

  const int orow0 = o0 + (hi ? 8 : 0);
  const int n = n0 + l;
  float* dst = y + (size_t)b * DIM * NPIX;
  #pragma unroll
  for (int r = 0; r < 8; ++r)
    dst[(size_t)(orow0 + r) * NPIX + n] = acc[r] + bo[orow0 + r];
}

// ---- launcher ---------------------------------------------------------------

extern "C" void kernel_launch(void* const* d_in, const int* in_sizes, int n_in,
                              void* d_out, int out_size, void* d_ws, size_t ws_size,
                              hipStream_t stream) {
  const float* x  = (const float*)d_in[0];
  const float* wk = (const float*)d_in[1];
  const float* bk = (const float*)d_in[2];
  const float* wq = (const float*)d_in[3];
  const float* bq = (const float*)d_in[4];
  const float* wv = (const float*)d_in[5];
  const float* bv = (const float*)d_in[6];
  const float* wo = (const float*)d_in[7];
  const float* bo = (const float*)d_in[8];
  float* out = (float*)d_out;

  char* ws = (char*)d_ws;
  const size_t szXT  = (size_t)BATCH * NPIX * DIM * sizeof(_Float16);          // 4.5 MB
  const size_t szWH  = (size_t)4 * DIM * DIM * sizeof(_Float16);               // 0.5 MB
  const size_t szHKV = (size_t)BATCH * HEADS * NPIX * HC * sizeof(_Float16);   // 4.5 MB

  _Float16* xT   = (_Float16*)(ws);
  _Float16* wh   = (_Float16*)(ws + szXT);
  _Float16* kT   = (_Float16*)(ws + szXT + szWH);
  _Float16* qT   = (_Float16*)(ws + szXT + szWH + szHKV);
  _Float16* vN   = (_Float16*)(ws + szXT + szWH + 2 * szHKV);
  _Float16* outT = (_Float16*)(ws + szXT + szWH + 3 * szHKV);

  // Stage 0: conversions
  cvt_x_kernel<<<(BATCH * NPIX * DIM) / 256, 256, 0, stream>>>(x, xT);
  cvt_w_kernel<<<(DIM * DIM) / 256, 256, 0, stream>>>(wk, wq, wv, wo, wh);

  // Stage 1: Q/K/V projections — 3*4*16*144 = 27648 wave-tiles, 8 waves/block
  proj_kernel<<<3456, 256, 0, stream>>>(xT, wh, bk, bq, bv, kT, qT, vN);

  // Stage 2: flash attention — 16 (b*h) x 18 m-blocks, 4 waves (128 queries)/block
  attn_kernel<<<288, 128, 0, stream>>>(qT, kT, vN, outT);

  // Stage 3: output projection — 4*16*144 = 9216 wave-tiles, 8 waves/block
  outproj_kernel<<<1152, 256, 0, stream>>>(wh, outT, bo, out);
}
